// HymbaHybridBlock_61246233641061
// MI455X (gfx1250) — matrix operand order repaired and checked by reading, not compiled, not run on hardware
//
#include <hip/hip_runtime.h>
#include <stdint.h>

#define B_   2
#define T_   2048
#define D_   2048
#define H_   16
#define KV_  4
#define Dh_  128
#define TK_  1152            // META(128) + window(1024)
#define SCALE_ 0.08838834764831845f   // 1/sqrt(128)

typedef __attribute__((ext_vector_type(16))) __bf16 v16bf;
typedef __attribute__((ext_vector_type(8)))  float  v8f;

union AFrag { v16bf v; unsigned u[8]; };

__device__ __forceinline__ uint16_t f2bf(float f) {
  unsigned u = __float_as_uint(f);
  u += 0x7FFFu + ((u >> 16) & 1u);       // round-to-nearest-even
  return (uint16_t)(u >> 16);
}

__device__ __forceinline__ v8f wmma_bf16(const v16bf& a, const v16bf& b, const v8f& c) {
  return __builtin_amdgcn_wmma_f32_16x16x32_bf16(false, a, false, b, (short)0, c, false, false);
}

// ---------------- prep: f32 -> bf16 convert / weight transpose ----------------
__global__ void cvt_x_kernel(const float* __restrict__ x, uint16_t* __restrict__ xb, int n) {
  int i = blockIdx.x * blockDim.x + threadIdx.x;
  if (i < n) xb[i] = f2bf(x[i]);
}

__global__ void transpose_w_kernel(const float* __restrict__ W, uint16_t* __restrict__ WT,
                                   int K, int N) {
  int i = blockIdx.x * blockDim.x + threadIdx.x;
  if (i >= K * N) return;
  int k = i / N, n = i % N;
  WT[(size_t)n * K + k] = f2bf(W[i]);
}

// ---------------- GEMM: C(MxN) = A(MxK,bf16) * BT(NxK,bf16)^T ----------------
// Each wave computes a 32x64 tile (2 A-frags, 8 WMMA accumulators, B reused 2x),
// K-loop step 32: 8 WMMAs per 6 fragment loads.
// mode 0: Q proj  (+RoPE) -> (B,H,T,Dh) bf16
// mode 1: K proj  (+RoPE) -> (B,KV,T,Dh) bf16
// mode 2: V proj          -> (B,KV,Dh,T) bf16 (transposed for PV WMMA)
// mode 3: out proj        -> f32 d_out (B*T, D)
__global__ void gemm_rope_kernel(const uint16_t* __restrict__ A,
                                 const uint16_t* __restrict__ BT,
                                 uint16_t* __restrict__ out16,
                                 float* __restrict__ out32,
                                 int M, int N, int Kd, int mode) {
  int lane = threadIdx.x & 31;
  int hf   = lane >> 4;
  int l16  = lane & 15;
  int gw   = (blockIdx.x * blockDim.x + threadIdx.x) >> 5;
  int tilesN = N >> 6;
  int mt = gw / tilesN;
  int nt = gw % tilesN;
  if (mt >= (M >> 5)) return;

  const uint16_t* Ar0 = A + (size_t)(mt * 32 + l16)      * Kd;
  const uint16_t* Ar1 = A + (size_t)(mt * 32 + 16 + l16) * Kd;

  v8f acc[2][4] = {};
  for (int k0 = 0; k0 < Kd; k0 += 32) {
    AFrag a0, a1;
    #pragma unroll
    for (int r = 0; r < 8; ++r) {
      int kk = k0 + ((r >> 2) << 4) + hf * 8 + ((r & 3) << 1);  // 16-bit A layout
      a0.u[r] = *(const unsigned*)(Ar0 + kk);
      a1.u[r] = *(const unsigned*)(Ar1 + kk);
    }
    __builtin_prefetch(Ar0 + k0 + 128, 0, 1);
    __builtin_prefetch(Ar1 + k0 + 128, 0, 1);
    #pragma unroll
    for (int j = 0; j < 4; ++j) {
      int col = nt * 64 + j * 16 + l16;
      const uint16_t* Br = BT + (size_t)col * Kd;
      AFrag b;
      #pragma unroll
      for (int r = 0; r < 8; ++r) {
        int kk = k0 + ((r >> 2) << 4) + hf * 8 + ((r & 3) << 1); // 16-bit B layout
        b.u[r] = *(const unsigned*)(Br + kk);
      }
      acc[0][j] = wmma_bf16(a0.v, b.v, acc[0][j]);
      acc[1][j] = wmma_bf16(a1.v, b.v, acc[1][j]);
    }
  }

  #pragma unroll
  for (int mi = 0; mi < 2; ++mi) {
    #pragma unroll
    for (int j = 0; j < 4; ++j) {
      #pragma unroll
      for (int r = 0; r < 8; ++r) {
        int mrow = mt * 32 + mi * 16 + r + 8 * hf;      // C layout: row = r + 8*half
        int col  = nt * 64 + j * 16 + l16;              //           col = lane%16
        int b    = mrow >> 11;
        int t    = mrow & (T_ - 1);
        float val = acc[mi][j][r];
        if (mode <= 1) {
          // RoPE: pairs (2i, 2i+1) are adjacent lanes in C layout
          float partner = __shfl_xor(val, 1);
          int nd = col & (Dh_ - 1);
          int i2 = nd >> 1;
          float inv = __expf(-(float)(2 * i2) * (9.210340371976184f / 128.0f)); // 10000^(-2i/128)
          float fr = (float)t * inv;
          float c = cosf(fr), s = sinf(fr);
          float o = ((col & 1) == 0) ? (val * c - partner * s)
                                     : (partner * s + val * c);
          if (mode == 0) {
            int h = col >> 7;
            out16[(((size_t)b * H_ + h) * T_ + t) * Dh_ + nd] = f2bf(o);
          } else {
            int kv = col >> 7;
            out16[(((size_t)b * KV_ + kv) * T_ + t) * Dh_ + nd] = f2bf(o);
          }
        } else if (mode == 2) {
          int kv = col >> 7;
          int nd = col & (Dh_ - 1);
          out16[(((size_t)b * KV_ + kv) * Dh_ + nd) * T_ + t] = f2bf(val);
        } else {
          out32[(size_t)mrow * N + col] = val;
        }
      }
    }
  }
}

// ---------------- flash attention over META + sliding window ----------------
// One wave per (b, h, 16-query tile). 32 keys per step: 8 QK^T WMMAs + 8 PV WMMAs.
__global__ void attn_kernel(const uint16_t* __restrict__ Q,
                            const uint16_t* __restrict__ Kc,
                            const uint16_t* __restrict__ Vt,
                            uint16_t* __restrict__ AO) {
  __shared__ uint16_t lds_p[4][16][32];
  int lane = threadIdx.x & 31;
  int w    = threadIdx.x >> 5;
  int hf   = lane >> 4, l16 = lane & 15;
  int gw   = blockIdx.x * 4 + w;
  int qt = gw & 127;
  int h  = (gw >> 7) & 15;
  int b  = gw >> 11;
  int kv = h >> 2;

  const uint16_t* qb = Q  + (((size_t)b * H_  + h ) * T_ ) * Dh_;
  const uint16_t* kb = Kc + (((size_t)b * KV_ + kv) * T_ ) * Dh_;
  const uint16_t* vb = Vt + (((size_t)b * KV_ + kv) * Dh_) * T_;

  // Q A-fragments kept resident (Dh=128 -> 4 chunks of K=32)
  AFrag qa[4];
  int trow = qt * 16 + l16;
  #pragma unroll
  for (int c = 0; c < 4; ++c)
    #pragma unroll
    for (int r = 0; r < 8; ++r) {
      int kk = c * 32 + ((r >> 2) << 4) + hf * 8 + ((r & 3) << 1);
      qa[c].u[r] = *(const unsigned*)(qb + (size_t)trow * Dh_ + kk);
    }

  v8f acc[8] = {};
  float Mr[8], Lr[8];
  #pragma unroll
  for (int r = 0; r < 8; ++r) { Mr[r] = -1e30f; Lr[r] = 0.0f; }

  int jmax   = qt * 16 + 15;                  // mask: concat index j <= query index i
  int nsteps = (jmax + 32) >> 5;
  if (nsteps > TK_ / 32) nsteps = TK_ / 32;

  for (int s = 0; s < nsteps; ++s) {
    int jbase = s * 32;
    int toff  = (jbase < 128) ? 0 : 896;      // j<128: meta t=j; else t=j+(T-1024-128)
    v8f s0 = {}, s1 = {};
    int krow0 = jbase + toff + l16;
    #pragma unroll
    for (int c = 0; c < 4; ++c) {
      AFrag b0, b1;
      #pragma unroll
      for (int r = 0; r < 8; ++r) {
        int kk = c * 32 + ((r >> 2) << 4) + hf * 8 + ((r & 3) << 1); // dh pairs contiguous
        b0.u[r] = *(const unsigned*)(kb + (size_t)krow0        * Dh_ + kk);
        b1.u[r] = *(const unsigned*)(kb + (size_t)(krow0 + 16) * Dh_ + kk);
      }
      s0 = wmma_bf16(qa[c].v, b0.v, s0);
      s1 = wmma_bf16(qa[c].v, b1.v, s1);
    }

    // scale + mask + online softmax (rows live across 16 lanes)
    float p0[8], p1[8], al[8];
    #pragma unroll
    for (int r = 0; r < 8; ++r) {
      int i  = qt * 16 + r + 8 * hf;
      int j0 = jbase + l16;
      float v0 = s0[r] * SCALE_;
      float v1 = s1[r] * SCALE_;
      if (j0      > i) v0 = -1e30f;
      if (j0 + 16 > i) v1 = -1e30f;
      float mx = fmaxf(v0, v1);
      mx = fmaxf(mx, __shfl_xor(mx, 1));
      mx = fmaxf(mx, __shfl_xor(mx, 2));
      mx = fmaxf(mx, __shfl_xor(mx, 4));
      mx = fmaxf(mx, __shfl_xor(mx, 8));
      float nM = fmaxf(Mr[r], mx);
      al[r] = __expf(Mr[r] - nM);
      Mr[r] = nM;
      p0[r] = __expf(v0 - nM);
      p1[r] = __expf(v1 - nM);
      float rs = p0[r] + p1[r];
      rs += __shfl_xor(rs, 1);
      rs += __shfl_xor(rs, 2);
      rs += __shfl_xor(rs, 4);
      rs += __shfl_xor(rs, 8);
      Lr[r] = Lr[r] * al[r] + rs;
    }
    #pragma unroll
    for (int dt = 0; dt < 8; ++dt)
      #pragma unroll
      for (int r = 0; r < 8; ++r)
        acc[dt][r] = acc[dt][r] * al[r];

    // C-layout P -> A-layout bf16 fragment via per-wave LDS bounce
    #pragma unroll
    for (int r = 0; r < 8; ++r) {
      int m = r + 8 * hf;
      lds_p[w][m][l16]      = f2bf(p0[r]);
      lds_p[w][m][16 + l16] = f2bf(p1[r]);
    }
    AFrag pa;
    #pragma unroll
    for (int r = 0; r < 8; ++r) {
      int kk = ((r >> 2) << 4) + hf * 8 + ((r & 3) << 1);
      pa.u[r] = *(const unsigned*)&lds_p[w][l16][kk];
    }

    // P @ V : V stored transposed (Dh,T) so key pairs are contiguous
    #pragma unroll
    for (int dt = 0; dt < 8; ++dt) {
      AFrag vf;
      #pragma unroll
      for (int r = 0; r < 8; ++r) {
        int tloc = jbase + toff + ((r >> 2) << 4) + hf * 8 + ((r & 3) << 1);
        vf.u[r] = *(const unsigned*)(vb + (size_t)(dt * 16 + l16) * T_ + tloc);
      }
      acc[dt] = wmma_bf16(pa.v, vf.v, acc[dt]);
    }
  }

  // normalize and emit (B,T,H*Dh) bf16 for output projection
  #pragma unroll
  for (int dt = 0; dt < 8; ++dt)
    #pragma unroll
    for (int r = 0; r < 8; ++r) {
      int m = r + 8 * hf;
      int t = qt * 16 + m;
      float o = acc[dt][r] / Lr[r];
      AO[((size_t)b * T_ + t) * (H_ * Dh_) + h * Dh_ + dt * 16 + l16] = f2bf(o);
    }
}

// ---------------- launch ----------------
extern "C" void kernel_launch(void* const* d_in, const int* in_sizes, int n_in,
                              void* d_out, int out_size, void* d_ws, size_t ws_size,
                              hipStream_t stream) {
  const float* x  = (const float*)d_in[0];
  const float* Wq = (const float*)d_in[1];
  const float* Wk = (const float*)d_in[2];
  const float* Wv = (const float*)d_in[3];
  const float* Wo = (const float*)d_in[4];
  char* ws = (char*)d_ws;

  uint16_t* xb  = (uint16_t*)(ws + 0);          // 16.78 MB  x as bf16
  uint16_t* WqT = (uint16_t*)(ws + 16777216);   //  8.39 MB  Wq^T bf16
  uint16_t* WkT = (uint16_t*)(ws + 25165824);   //  2.10 MB
  uint16_t* WvT = (uint16_t*)(ws + 27262976);   //  2.10 MB
  uint16_t* WoT = (uint16_t*)(ws + 29360128);   //  8.39 MB
  uint16_t* Qb  = (uint16_t*)(ws + 37748736);   // 16.78 MB  (B,H,T,Dh)
  uint16_t* Kb  = (uint16_t*)(ws + 54525952);   //  4.19 MB  (B,KV,T,Dh)
  uint16_t* Vtb = (uint16_t*)(ws + 58720256);   //  4.19 MB  (B,KV,Dh,T)
  uint16_t* AO  = (uint16_t*)(ws + 62914560);   // 16.78 MB  attn out bf16
  float* out = (float*)d_out;

  int nx = B_ * T_ * D_;
  cvt_x_kernel<<<(nx + 255) / 256, 256, 0, stream>>>(x, xb, nx);
  transpose_w_kernel<<<(D_ * 2048 + 255) / 256, 256, 0, stream>>>(Wq, WqT, D_, 2048);
  transpose_w_kernel<<<(D_ * 512  + 255) / 256, 256, 0, stream>>>(Wk, WkT, D_, 512);
  transpose_w_kernel<<<(D_ * 512  + 255) / 256, 256, 0, stream>>>(Wv, WvT, D_, 512);
  transpose_w_kernel<<<(2048 * D_ + 255) / 256, 256, 0, stream>>>(Wo, WoT, 2048, D_);

  int M = B_ * T_;
  {
    int waves = (M / 32) * (2048 / 64);
    gemm_rope_kernel<<<waves / 4, 128, 0, stream>>>(xb, WqT, Qb, nullptr, M, 2048, D_, 0);
  }
  {
    int waves = (M / 32) * (512 / 64);
    gemm_rope_kernel<<<waves / 4, 128, 0, stream>>>(xb, WkT, Kb, nullptr, M, 512, D_, 1);
  }
  {
    int waves = (M / 32) * (512 / 64);
    gemm_rope_kernel<<<waves / 4, 128, 0, stream>>>(xb, WvT, Vtb, nullptr, M, 512, D_, 2);
  }

  attn_kernel<<<(B_ * H_ * (T_ / 16)) / 4, 128, 0, stream>>>(Qb, Kb, Vtb, AO);

  {
    int waves = (M / 32) * (2048 / 64);
    gemm_rope_kernel<<<waves / 4, 128, 0, stream>>>(AO, WoT, nullptr, out, M, 2048, D_, 3);
  }
}
